// TriangleAttentionBlock_31155692765638
// MI455X (gfx1250) — matrix operand order repaired and checked by reading, not compiled, not run on hardware
//
#include <hip/hip_runtime.h>

// ---------------------------------------------------------------------------
// TriangleAttentionBlock for MI455X (gfx1250, wave32, WMMA)
// Flash-attention fusion; all WMMA operands are contiguous 128-bit loads.
//   Pre:     x -> f16, W -> f16 transposed [col][k]
//   Phase 1: QKV projections (v_wmma_f32_16x16x32_f16), V stored transposed
//   Phase 2: one wave = (head, 32-query tile); K/V fragments shared by two
//            16-row score/PV WMMA pairs; output pooled over j in registers,
//            flushed with 2 global_atomic_add_f32 per lane.
// ---------------------------------------------------------------------------

typedef __attribute__((ext_vector_type(16))) _Float16 v16h;
typedef __attribute__((ext_vector_type(8)))  _Float16 v8h;
typedef __attribute__((ext_vector_type(8)))  float    v8f;

#define S_TOTAL 4096   // N_ATOMS * N_ATOMS
#define DIN     128
#define NH      4
#define DH      32
#define NA      64

__device__ __forceinline__ v8f wmma_f16(v16h a, v16h b, v8f c) {
  // D = A(16x32 f16) * B(32x16 f16) + C(16x16 f32)
  return __builtin_amdgcn_wmma_f32_16x16x32_f16(
      /*neg_a=*/false, a, /*neg_b=*/false, b,
      /*c_mod=*/(short)0, c, /*reuse_a=*/false, /*reuse_b=*/false);
}

__device__ __forceinline__ v16h cat8(v8h lo, v8h hi) {
  return __builtin_shufflevector(lo, hi, 0, 1, 2, 3, 4, 5, 6, 7,
                                 8, 9, 10, 11, 12, 13, 14, 15);
}

// Online-softmax update for one 16x32 score block held as two 16x16 C tiles.
// Row stats live per-lane for the 8 rows this lane's C half touches.
__device__ __forceinline__ void online_softmax(const v8f& c0, const v8f& c1,
                                               int valid0, int valid1,
                                               float Mrow[8], float Lrow[8],
                                               v8f& o0, v8f& o1,
                                               float p0[8], float p1[8]) {
  const float scale = 0.17677669529663689f;        // 1/sqrt(32)
#pragma unroll
  for (int r = 0; r < 8; ++r) {
    float sc0 = valid0 ? c0[r] * scale : -3.0e38f;
    float sc1 = valid1 ? c1[r] * scale : -3.0e38f;
    float cm = fmaxf(sc0, sc1);                    // row max over 16 lanes
    cm = fmaxf(cm, __shfl_xor(cm, 1, 32));
    cm = fmaxf(cm, __shfl_xor(cm, 2, 32));
    cm = fmaxf(cm, __shfl_xor(cm, 4, 32));
    cm = fmaxf(cm, __shfl_xor(cm, 8, 32));
    float mnew  = fmaxf(Mrow[r], cm);
    float alpha = __expf(Mrow[r] - mnew);
    float e0 = __expf(sc0 - mnew);
    float e1 = __expf(sc1 - mnew);
    float rs = e0 + e1;
    rs += __shfl_xor(rs, 1, 32);
    rs += __shfl_xor(rs, 2, 32);
    rs += __shfl_xor(rs, 4, 32);
    rs += __shfl_xor(rs, 8, 32);
    Lrow[r] = Lrow[r] * alpha + rs;
    Mrow[r] = mnew;
    o0[r] *= alpha;
    o1[r] *= alpha;
    p0[r] = e0;
    p1[r] = e1;
  }
}

__global__ void zero_kernel(float* __restrict__ out, int n) {
  int i = blockIdx.x * blockDim.x + threadIdx.x;
  if (i < n) out[i] = 0.0f;
}

// x (S,128) f32 -> f16, same layout
__global__ void cvt_x_kernel(const float* __restrict__ x, _Float16* __restrict__ xh) {
  int i = blockIdx.x * blockDim.x + threadIdx.x;   // < S_TOTAL*DIN
  xh[i] = (_Float16)x[i];
}

// W (128,128) f32 -> Wt f16 [mat][col][k]  (transposed for contiguous B frags)
__global__ void cvt_w_kernel(const float* __restrict__ Wq,
                             const float* __restrict__ Wk,
                             const float* __restrict__ Wv,
                             _Float16* __restrict__ wt) {
  int i = blockIdx.x * blockDim.x + threadIdx.x;   // < 3*128*128
  int w   = i >> 14;
  int rem = i & 16383;
  int col = rem >> 7;
  int k   = rem & 127;
  const float* W = (w == 0) ? Wq : (w == 1) ? Wk : Wv;
  wt[i] = (_Float16)W[k * DIN + col];
}

// ---------------------------------------------------------------------------
// Phase 1: Y = X @ W + b. One wave = one 16x16 tile, K=128 as 4 WMMAs.
// Units: 3 mats * 256 mtiles * 8 ntiles = 6144 waves.
// Q,K stored [h][s][d]; V stored transposed [h][d][t].
// ---------------------------------------------------------------------------
__global__ void qkv_kernel(const _Float16* __restrict__ xh,
                           const _Float16* __restrict__ wt,
                           const float* __restrict__ bq,
                           const float* __restrict__ bk,
                           const float* __restrict__ bv,
                           _Float16* __restrict__ qkv) {
  const int wave = threadIdx.x >> 5;
  const int lane = threadIdx.x & 31;
  const int unit = blockIdx.x * 4 + wave;          // < 6144
  const int w     = unit >> 11;                    // 0..2 (q/k/v)
  const int rem   = unit & 2047;
  const int mtile = rem >> 3;                      // 0..255
  const int ntile = rem & 7;                       // 0..7

  const float* bias = (w == 0) ? bq : (w == 1) ? bk : bv;

  const int halfsel = lane >> 4;                   // ISA 16-bit operand layout
  const int lx      = lane & 15;
  const int a_kbase = halfsel * 8;                 // A: K 0-7 / 8-15 per half
  const int b_kbase = halfsel * 8;                 // B: K 0-7 / 8-15 per half

  const _Float16* arow = xh + (size_t)(mtile * 16 + lx) * DIN;           // lane=M
  const _Float16* brow = wt + ((size_t)w * DIN + ntile * 16 + lx) * DIN; // lane=N

  v8f c = {};
#pragma unroll
  for (int kb = 0; kb < DIN; kb += 32) {
    v16h a = cat8(*(const v8h*)(arow + kb + a_kbase),
                  *(const v8h*)(arow + kb + 16 + a_kbase));
    v16h b = cat8(*(const v8h*)(brow + kb + b_kbase),
                  *(const v8h*)(brow + kb + 16 + b_kbase));
    c = wmma_f16(a, b, c);
  }

  // C layout: lane = N col, VGPR r -> M = r (lanes 0-15) / r+8 (lanes 16-31)
  const int col = ntile * 16 + lx;
  const int h   = col >> 5;
  const int d   = col & 31;
  const float bb = bias[col];
  if (w < 2) {                                      // Q,K: [h][s][d]
#pragma unroll
    for (int r = 0; r < 8; ++r) {
      int s = mtile * 16 + (halfsel ? (r + 8) : r);
      qkv[(((size_t)w * NH + h) * S_TOTAL + s) * DH + d] = (_Float16)(c[r] + bb);
    }
  } else {                                          // V: [h][d][t] (transposed)
    _Float16* vt = qkv + 2 * (size_t)NH * S_TOTAL * DH +
                   ((size_t)h * DH + d) * S_TOTAL + mtile * 16 + halfsel * 8;
#pragma unroll
    for (int r = 0; r < 8; ++r)                     // consecutive t: packs
      vt[r] = (_Float16)(c[r] + bb);
  }
}

// ---------------------------------------------------------------------------
// Phase 2: flash attention. One wave = (head, 32-query tile): 512 units.
// 32 keys/iter: K/V fragments shared by both 16-row halves of the query tile.
// ---------------------------------------------------------------------------
__global__ void attn_kernel(const _Float16* __restrict__ qkv,
                            const int* __restrict__ atom_mask,
                            float* __restrict__ out) {
  __shared__ __align__(16) _Float16 smem[4 * 2 * 16 * 32]; // 2 P tiles / wave
  const int wave = threadIdx.x >> 5;
  const int lane = threadIdx.x & 31;
  _Float16* sm0 = smem + wave * 1024;
  _Float16* sm1 = sm0 + 512;

  const int unit = blockIdx.x * 4 + wave;          // < 512
  const int h    = unit >> 7;                      // head
  const int qt   = unit & 127;
  const int s0   = qt * 32;

  const _Float16* qws = qkv;                                   // [h][s][d]
  const _Float16* kws = qkv + (size_t)NH * S_TOTAL * DH;       // [h][s][d]
  const _Float16* vws = qkv + 2 * (size_t)NH * S_TOTAL * DH;   // [h][d][t]

  const int halfsel = lane >> 4;
  const int lx      = lane & 15;
  const int a_kbase = halfsel * 8;
  const int b_kbase = halfsel * 8;

  // Two Q row-tiles as A fragments (constant over key loop)
  v16h aq0, aq1;
  {
    const _Float16* q0 = qws + ((size_t)h * S_TOTAL + (s0 + lx)) * DH;
    const _Float16* q1 = q0 + 16 * DH;
    aq0 = cat8(*(const v8h*)(q0 + a_kbase), *(const v8h*)(q0 + 16 + a_kbase));
    aq1 = cat8(*(const v8h*)(q1 + a_kbase), *(const v8h*)(q1 + 16 + a_kbase));
  }

  float M0[8], L0[8], M1[8], L1[8], p00[8], p01[8], p10[8], p11[8];
  v8f o00 = {}, o01 = {}, o10 = {}, o11 = {};      // [qtile][dhalf]
#pragma unroll
  for (int r = 0; r < 8; ++r) {
    M0[r] = -3.0e38f; L0[r] = 0.0f;
    M1[r] = -3.0e38f; L1[r] = 0.0f;
  }

  for (int kb = 0; kb < S_TOTAL; kb += 32) {
    const int t0 = kb + lx, t1 = kb + 16 + lx;
    const _Float16* krow0 = kws + ((size_t)h * S_TOTAL + t0) * DH + b_kbase;
    const _Float16* krow1 = kws + ((size_t)h * S_TOTAL + t1) * DH + b_kbase;
    if (kb + 64 < S_TOTAL) {                        // stage next K tile
      __builtin_prefetch((const void*)(krow0 + 32 * DH), 0, 1);
      __builtin_prefetch((const void*)(krow1 + 32 * DH), 0, 1);
    }

    // K fragments (two 8-half runs 16 apart -> 2x global_load_b128 each),
    // shared by both query row-tiles.
    v16h bk0 = cat8(*(const v8h*)krow0, *(const v8h*)(krow0 + 16));
    v16h bk1 = cat8(*(const v8h*)krow1, *(const v8h*)(krow1 + 16));
    v8f c00 = {}, c01 = {}, c10 = {}, c11 = {};
    c00 = wmma_f16(aq0, bk0, c00);                  // qrows 0-15,  keys kb+0..15
    c01 = wmma_f16(aq0, bk1, c01);                  // qrows 0-15,  keys kb+16..31
    c10 = wmma_f16(aq1, bk0, c10);                  // qrows 16-31, keys kb+0..15
    c11 = wmma_f16(aq1, bk1, c11);                  // qrows 16-31, keys kb+16..31

    const int valid0 = atom_mask[t0 >> 6] & atom_mask[t0 & 63];
    const int valid1 = atom_mask[t1 >> 6] & atom_mask[t1 & 63];

    online_softmax(c00, c01, valid0, valid1, M0, L0, o00, o01, p00, p01);
    online_softmax(c10, c11, valid0, valid1, M1, L1, o10, o11, p10, p11);

    // P: C layout (lane = key col) -> LDS -> A layout (lane = query row)
#pragma unroll
    for (int r = 0; r < 8; ++r) {
      int m = halfsel ? (r + 8) : r;
      sm0[m * 32 + lx]      = (_Float16)p00[r];
      sm0[m * 32 + 16 + lx] = (_Float16)p01[r];
      sm1[m * 32 + lx]      = (_Float16)p10[r];
      sm1[m * 32 + 16 + lx] = (_Float16)p11[r];
    }
    __syncthreads();                                // uniform trip count: safe

    v16h ap0 = cat8(*(const v8h*)(sm0 + lx * 32 + a_kbase),      // ds_load_b128
                    *(const v8h*)(sm0 + lx * 32 + 16 + a_kbase));
    v16h ap1 = cat8(*(const v8h*)(sm1 + lx * 32 + a_kbase),
                    *(const v8h*)(sm1 + lx * 32 + 16 + a_kbase));

    // V fragments (transposed [h][d][t]): shared by both query row-tiles
    const _Float16* vb = vws + (size_t)h * DH * S_TOTAL + kb + b_kbase;
    v16h bv0 = cat8(*(const v8h*)(vb + (size_t)lx * S_TOTAL),
                    *(const v8h*)(vb + (size_t)lx * S_TOTAL + 16));        // d = lx
    v16h bv1 = cat8(*(const v8h*)(vb + (size_t)(16 + lx) * S_TOTAL),
                    *(const v8h*)(vb + (size_t)(16 + lx) * S_TOTAL + 16)); // d = 16+lx
    o00 = wmma_f16(ap0, bv0, o00);
    o01 = wmma_f16(ap0, bv1, o01);
    o10 = wmma_f16(ap1, bv0, o10);
    o11 = wmma_f16(ap1, bv1, o11);
  }

  // All 32 rows of this tile share the same output row i = s0>>6 and the
  // lane's output column: pool over rows in registers, then ONE atomic per
  // (lane, d-half).
  const int i = s0 >> 6;
  float acc0 = 0.0f, acc1 = 0.0f;
#pragma unroll
  for (int r = 0; r < 8; ++r) {
    float inv0 = (L0[r] > 0.0f) ? (1.0f / L0[r]) : 0.0f;
    float inv1 = (L1[r] > 0.0f) ? (1.0f / L1[r]) : 0.0f;
    acc0 += o00[r] * inv0 + o10[r] * inv1;
    acc1 += o01[r] * inv0 + o11[r] * inv1;
  }
  atomicAdd(&out[i * (NH * DH) + h * DH + lx],      acc0);
  atomicAdd(&out[i * (NH * DH) + h * DH + 16 + lx], acc1);
}

// ---------------------------------------------------------------------------
extern "C" void kernel_launch(void* const* d_in, const int* in_sizes, int n_in,
                              void* d_out, int out_size, void* d_ws, size_t ws_size,
                              hipStream_t stream) {
  const float* x    = (const float*)d_in[0];   // (64,64,128)
  const int*   mask = (const int*)d_in[1];     // (64,)
  const float* Wq   = (const float*)d_in[2];
  const float* bq   = (const float*)d_in[3];
  const float* Wk   = (const float*)d_in[4];
  const float* bk   = (const float*)d_in[5];
  const float* Wv   = (const float*)d_in[6];
  const float* bv   = (const float*)d_in[7];
  float* out = (float*)d_out;                  // (64,128) f32

  // workspace layout (f16 elems):
  //   [0)            qkv: q[h][s][d], k[h][s][d], v[h][d][t]  (3*4*4096*32)
  //   [1572864)      xh : x in f16                            (4096*128)
  //   [2097152)      wt : W^T f16 [mat][col][k]               (3*128*128)
  _Float16* qkv = (_Float16*)d_ws;
  _Float16* xh  = qkv + 3 * (size_t)NH * S_TOTAL * DH;
  _Float16* wt  = xh + (size_t)S_TOTAL * DIN;

  zero_kernel<<<(out_size + 255) / 256, 256, 0, stream>>>(out, out_size);
  cvt_x_kernel<<<(S_TOTAL * DIN) / 256, 256, 0, stream>>>(x, xh);
  cvt_w_kernel<<<(3 * DIN * DIN) / 256, 256, 0, stream>>>(Wq, Wk, Wv, wt);
  qkv_kernel<<<6144 / 4, 128, 0, stream>>>(xh, wt, bq, bk, bv, qkv);
  attn_kernel<<<512 / 4, 128, 0, stream>>>(qkv, mask, out);
}
